// RotEncoderMLP_25718264168603
// MI455X (gfx1250) — matrix operand order repaired and checked by reading, not compiled, not run on hardware
//
#include <hip/hip_runtime.h>
#include <hip/hip_bf16.h>
#include <stdint.h>

// ---------------------------------------------------------------------------
// RotEncoderMLP for MI455X (gfx1250): bf16 WMMA 16x16x32 GEMM pipeline with
// CDNA5 data movement: TDM tensor_load_to_lds for GEMM2 tiles, async
// global->LDS loads for GEMM3 B tiles.
// ---------------------------------------------------------------------------

#define R_ROWS   10240          // B*N
#define HID      2048
#define OUTD     512
#define KDIM     8192           // 4*HID
#define M_TOTAL  40960          // 4*R_ROWS

#define MT 128                  // GEMM M tile
#define NT 128                  // GEMM N tile
#define KT 32                   // WMMA K
#define LDSPITCH 80             // bytes per 32-bf16 LDS row (64B data + 16B pad)

typedef __attribute__((ext_vector_type(16))) __bf16 v16bf;
typedef __attribute__((ext_vector_type(8)))  float  v8f;
typedef unsigned int u32x4 __attribute__((ext_vector_type(4)));
typedef int          i32x4 __attribute__((ext_vector_type(4)));
typedef int          i32x8 __attribute__((ext_vector_type(8)));

#define AS1 __attribute__((address_space(1)))
#define AS3 __attribute__((address_space(3)))

// Feature detection (compile-safe fallbacks)
#if defined(__has_builtin)
#  if __has_builtin(__builtin_amdgcn_tensor_load_to_lds)
#    define HAVE_TDM 1
#  else
#    define HAVE_TDM 0
#  endif
#  if __has_builtin(__builtin_amdgcn_global_load_async_to_lds_b128)
#    define HAVE_ASYNC_LDS 1
#  else
#    define HAVE_ASYNC_LDS 0
#  endif
#else
#  define HAVE_TDM 0
#  define HAVE_ASYNC_LDS 0
#endif

union Frag { uint4 u[2]; v16bf v; };

__device__ __forceinline__ unsigned short f2bf(float x) {
    unsigned int u = __float_as_uint(x);
    unsigned int r = ((u >> 16) & 1u) + 0x7FFFu;   // round-to-nearest-even
    return (unsigned short)((u + r) >> 16);
}
__device__ __forceinline__ float bf2f(unsigned short h) {
    return __uint_as_float(((unsigned int)h) << 16);
}

__device__ __forceinline__ void wait_asynccnt0() {
#if defined(__has_builtin) && __has_builtin(__builtin_amdgcn_s_wait_asynccnt)
    __builtin_amdgcn_s_wait_asynccnt(0);
#else
    asm volatile("s_wait_asynccnt 0x0" ::: "memory");
#endif
}

#if HAVE_ASYNC_LDS
// Builtin signature (from compiler diagnostic): param0 is `int4 __device__*`
// (addrspace(1)), so: (int4 AS1* gsrc, int4 AS3* ldst, imm offset, imm cpol).
// LDS byte offset = low 32 bits of the generic LDS address.
__device__ __forceinline__ void async_b128(const void* g, void* l) {
    __builtin_amdgcn_global_load_async_to_lds_b128(
        (AS1 i32x4*)(unsigned long long)g,
        (AS3 i32x4*)(unsigned int)(unsigned long long)l,
        0, 0);
}
#endif

#if HAVE_TDM
// 2D TDM tile load: `rows` rows of 32 bf16 (64B) each, row stride
// `row_stride_elems` bf16 in memory; LDS rows padded 64B data + 16B pad
// (pad_interval code 3 = 16 DWORDs, pad_amount code 3 = 4 DWORDs) -> 80B pitch.
__device__ __forceinline__ void tdm_load_tile_2d(unsigned lds_off,
                                                 const void* gsrc,
                                                 unsigned rows,
                                                 unsigned long long row_stride_elems) {
    unsigned long long ga = (unsigned long long)gsrc;
    u32x4 g0;
    g0[0] = 0x1u;                                           // count=1, user D#
    g0[1] = lds_off;                                        // lds_addr
    g0[2] = (unsigned)ga;                                   // global_addr lo
    g0[3] = (unsigned)((ga >> 32) & 0x01FFFFFFull) | (2u << 30);  // addr hi | type=2
    const unsigned td0 = (unsigned)row_stride_elems;        // tensor_dim0
    const unsigned td1 = rows;                              // tensor_dim1
    i32x8 g1;
    g1[0] = (int)((1u << 16)        // data_size = 2 bytes
                | (1u << 20)        // pad_enable
                | (3u << 22)        // pad_interval: 16 DWORDs
                | (3u << 25));      // pad_amount:   4 DWORDs
    g1[1] = (int)((td0 & 0xFFFFu) << 16);                   // dim0[15:0] (bar addr=0)
    g1[2] = (int)((td0 >> 16) | ((td1 & 0xFFFFu) << 16));   // dim0[31:16] | dim1[15:0]
    g1[3] = (int)((td1 >> 16) | (32u << 16));               // dim1[31:16] | tile_dim0=32
    g1[4] = (int)rows;                                      // tile_dim1, tile_dim2=0
    g1[5] = (int)(unsigned)(row_stride_elems & 0xFFFFFFFFull);      // dim0_stride lo
    g1[6] = (int)(unsigned)((row_stride_elems >> 32) & 0xFFFFull);  // dim0_stride hi
    g1[7] = 0;                                              // dim1_stride unused (2D)
    i32x4 z4 = {0, 0, 0, 0};
#if __clang_major__ >= 23
    i32x8 z8 = {0, 0, 0, 0, 0, 0, 0, 0};
    __builtin_amdgcn_tensor_load_to_lds(g0, g1, z4, z4, z8, 0);
#else
    __builtin_amdgcn_tensor_load_to_lds(g0, g1, z4, z4, 0);
#endif
}
#endif  // HAVE_TDM

// ---------------------------------------------------------------------------
// Weight prep: w2 (4,2048,2048) f32 -> w2t bf16 [i][d*2048+j]
// ---------------------------------------------------------------------------
__global__ void prep_w2_kernel(const float* __restrict__ w2,
                               unsigned short* __restrict__ w2t) {
    size_t idx = (size_t)blockIdx.x * 256 + threadIdx.x;   // 16,777,216 total
    int d = (int)(idx >> 22);
    int i = (int)((idx >> 11) & 2047);
    int j = (int)(idx & 2047);
    w2t[(size_t)i * KDIM + d * HID + j] = f2bf(w2[idx]);
}

__global__ void prep_w3_kernel(const float* __restrict__ w3,
                               unsigned short* __restrict__ w3t) {
    size_t idx = (size_t)blockIdx.x * 256 + threadIdx.x;   // 4,194,304 total
    int d = (int)(idx >> 20);
    int i = (int)((idx >> 11) & 511);
    int j = (int)(idx & 2047);
    w3t[(size_t)i * KDIM + d * HID + j] = f2bf(w3[idx]);
}

// ---------------------------------------------------------------------------
// Layer 1: orbit stack + c4conv(w1,b1) + relu -> h1 bf16 [r][g][i]
// ---------------------------------------------------------------------------
__global__ __launch_bounds__(256) void layer1_kernel(
    const float* __restrict__ ins,   // [R][5][5]
    const float* __restrict__ w1,    // [4][HID][7]
    const float* __restrict__ b1,    // [HID]
    unsigned short* __restrict__ h1) // [R][4][HID] bf16
{
    __shared__ float xs[16][4][7];
    const int tid = threadIdx.x;
    const int r0  = blockIdx.x * 16;

    if (tid < 16) {
        const float* p = ins + (size_t)(r0 + tid) * 25;
        float M[5][5];
        #pragma unroll
        for (int a = 0; a < 5; a++)
            #pragma unroll
            for (int b = 0; b < 5; b++) M[a][b] = p[a * 5 + b];
        // rot^h(M)[r][c]: h=0:(r,c) h=1:(c,4-r) h=2:(4-r,4-c) h=3:(4-c,r)
        #pragma unroll
        for (int h = 0; h < 4; h++) {
            xs[tid][h][0] = M[2][2];
            #pragma unroll
            for (int rr = 0; rr < 2; rr++)
                #pragma unroll
                for (int cc = 0; cc < 3; cc++) {
                    int sr, sc;
                    if      (h == 0) { sr = rr;     sc = cc;     }
                    else if (h == 1) { sr = cc;     sc = 4 - rr; }
                    else if (h == 2) { sr = 4 - rr; sc = 4 - cc; }
                    else             { sr = 4 - cc; sc = rr;     }
                    xs[tid][h][1 + rr * 3 + cc] = M[sr][sc];
                }
        }
    }
    __syncthreads();

    for (int q = 0; q < 8; q++) {
        const int i = tid + q * 256;
        float w[4][7];
        #pragma unroll
        for (int d = 0; d < 4; d++)
            #pragma unroll
            for (int j = 0; j < 7; j++)
                w[d][j] = w1[((size_t)d * HID + i) * 7 + j];
        const float bias = b1[i];
        #pragma unroll 4
        for (int row = 0; row < 16; row++) {
            #pragma unroll
            for (int g = 0; g < 4; g++) {
                float acc = bias;
                #pragma unroll
                for (int d = 0; d < 4; d++) {
                    const int h = (g + d) & 3;
                    #pragma unroll
                    for (int j = 0; j < 7; j++) acc += w[d][j] * xs[row][h][j];
                }
                acc = fmaxf(acc, 0.0f);
                h1[(size_t)(r0 + row) * KDIM + g * HID + i] = f2bf(acc);
            }
        }
    }
}

// ---------------------------------------------------------------------------
// GEMM2: h2[r][g][i] = bf16( bias2[i] + sum_k A[(g,r),k] * w2t[i][k] )
// A[(g,r), d*HID+j] = h1[r][(g+d)&3][j]. Tiles staged by TDM (preferred).
// ---------------------------------------------------------------------------
__global__ __launch_bounds__(256) void gemm2_kernel(
    const unsigned short* __restrict__ h1,   // [R][4][HID] bf16
    const unsigned short* __restrict__ w2t,  // [HID][KDIM] bf16
    const float* __restrict__ b2,            // [HID]
    unsigned short* __restrict__ h2)         // [R][4][HID] bf16
{
    __shared__ __align__(16) unsigned char lds[2 * MT * LDSPITCH];
    unsigned char* ldsA = lds;
    unsigned char* ldsB = lds + MT * LDSPITCH;

    const int tid  = threadIdx.x;
    const int lane = tid & 31;
    const int wv   = tid >> 5;
    const int wm   = wv & 3;
    const int wn   = wv >> 2;
    const int half = lane >> 4;
    const int l16  = lane & 15;

    const int bm = blockIdx.x;                // 0..319  (g-major M tiles)
    const int g  = bm / (R_ROWS / MT);        // 80 tiles per group
    const int r0 = (bm % (R_ROWS / MT)) * MT;
    const int n0 = blockIdx.y * NT;

#if HAVE_TDM
    const unsigned ldsA_off = (unsigned)(unsigned long long)(void*)ldsA;
    const unsigned ldsB_off = (unsigned)(unsigned long long)(void*)ldsB;
#endif

    v8f acc[2][4];
    #pragma unroll
    for (int a = 0; a < 2; a++)
        #pragma unroll
        for (int b = 0; b < 4; b++)
            #pragma unroll
            for (int e = 0; e < 8; e++) acc[a][b][e] = 0.0f;

    for (int kt = 0; kt < KDIM / KT; ++kt) {
        const int k0 = kt * KT;
        const int d  = k0 >> 11;
        const int j0 = k0 & (HID - 1);
        const int hs = (g + d) & 3;

#if HAVE_TDM
        // ---- TDM: two descriptor loads cover the whole A and B tiles ----
        if (wv == 0) {
            tdm_load_tile_2d(ldsA_off, h1 + (size_t)r0 * KDIM + hs * HID + j0,
                             MT, KDIM);
            tdm_load_tile_2d(ldsB_off, w2t + (size_t)n0 * KDIM + k0,
                             NT, KDIM);
            __builtin_amdgcn_s_wait_tensorcnt(0);
        }
#elif HAVE_ASYNC_LDS
        #pragma unroll
        for (int c = 0; c < 2; c++) {
            const int chunk = tid * 2 + c;
            const int row   = chunk >> 2;
            const int cp    = chunk & 3;
            async_b128(h1 + (size_t)(r0 + row) * KDIM + hs * HID + j0 + cp * 8,
                       ldsA + row * LDSPITCH + cp * 16);
            async_b128(w2t + (size_t)(n0 + row) * KDIM + k0 + cp * 8,
                       ldsB + row * LDSPITCH + cp * 16);
        }
        wait_asynccnt0();
#else
        #pragma unroll
        for (int c = 0; c < 2; c++) {
            const int chunk = tid * 2 + c;
            const int row   = chunk >> 2;
            const int cp    = chunk & 3;
            const unsigned short* srcA =
                h1 + (size_t)(r0 + row) * KDIM + hs * HID + j0 + cp * 8;
            *(uint4*)(ldsA + row * LDSPITCH + cp * 16) = *(const uint4*)srcA;
            const unsigned short* srcB =
                w2t + (size_t)(n0 + row) * KDIM + k0 + cp * 8;
            *(uint4*)(ldsB + row * LDSPITCH + cp * 16) = *(const uint4*)srcB;
        }
#endif
        __syncthreads();

        // ---- build fragments (ISA 7.12.2 16-bit layouts) ----
        Frag af[2], bfr[4];
        #pragma unroll
        for (int mi = 0; mi < 2; mi++) {
            const int row = wm * 32 + mi * 16 + l16;
            const unsigned char* p = ldsA + row * LDSPITCH + half * 16;
            af[mi].u[0] = *(const uint4*)(p);        // K 0-7   / 8-15
            af[mi].u[1] = *(const uint4*)(p + 32);   // K 16-23 / 24-31
        }
        #pragma unroll
        for (int ni = 0; ni < 4; ni++) {
            const int col = wn * 64 + ni * 16 + l16;
            const unsigned char* p = ldsB + col * LDSPITCH + half * 16;
            bfr[ni].u[0] = *(const uint4*)(p);       // K 0-7   / 8-15
            bfr[ni].u[1] = *(const uint4*)(p + 32);  // K 16-23 / 24-31
        }
        #pragma unroll
        for (int mi = 0; mi < 2; mi++)
            #pragma unroll
            for (int ni = 0; ni < 4; ni++)
                acc[mi][ni] = __builtin_amdgcn_wmma_f32_16x16x32_bf16(
                    false, af[mi].v, false, bfr[ni].v,
                    (short)0, acc[mi][ni], false, false);
        __syncthreads();
    }

    // ---- epilogue: + bias (relu comes after LN, not here), bf16 ----
    #pragma unroll
    for (int ni = 0; ni < 4; ni++) {
        const int col  = n0 + wn * 64 + ni * 16 + l16;
        const float bias = b2[col];
        #pragma unroll
        for (int mi = 0; mi < 2; mi++) {
            const int mbase = wm * 32 + mi * 16 + half * 8;
            #pragma unroll
            for (int v = 0; v < 8; v++) {
                const int r = r0 + mbase + v;
                h2[(size_t)r * KDIM + g * HID + col] = f2bf(acc[mi][ni][v] + bias);
            }
        }
    }
}

// ---------------------------------------------------------------------------
// LayerNorm stats: one block per (r,h) row of 2048 bf16 values.
// ---------------------------------------------------------------------------
__global__ __launch_bounds__(256) void ln_stats_kernel(
    const unsigned short* __restrict__ h2,  // [M_TOTAL][2048] (row = r*4+h)
    float* __restrict__ mean,
    float* __restrict__ rstd)
{
    __shared__ float rs[256], rs2[256];
    const int tid = threadIdx.x;
    const unsigned short* p = h2 + (size_t)blockIdx.x * HID;
    float s = 0.0f, s2 = 0.0f;
    for (int j = tid; j < HID; j += 256) {
        const float v = bf2f(p[j]);
        s += v; s2 += v * v;
    }
    rs[tid] = s; rs2[tid] = s2;
    __syncthreads();
    for (int o = 128; o > 0; o >>= 1) {
        if (tid < o) { rs[tid] += rs[tid + o]; rs2[tid] += rs2[tid + o]; }
        __syncthreads();
    }
    if (tid == 0) {
        const float m   = rs[0] * (1.0f / HID);
        const float var = rs2[0] * (1.0f / HID) - m * m;
        mean[blockIdx.x] = m;
        rstd[blockIdx.x] = rsqrtf(var + 1e-5f);
    }
}

// ---------------------------------------------------------------------------
// GEMM3: out[r][g][o] = b3[o] + sum_k relu(LN(h2))[(g,r),k] * w3t[o][k]
// LN+relu fused into A staging (VALU); B staged with async global->LDS.
// ---------------------------------------------------------------------------
__global__ __launch_bounds__(256) void gemm3_kernel(
    const unsigned short* __restrict__ h2,   // [R][4][HID] bf16
    const unsigned short* __restrict__ w3t,  // [OUTD][KDIM] bf16
    const float* __restrict__ b3,            // [OUTD]
    const float* __restrict__ mean,          // [M_TOTAL]  (r*4+h)
    const float* __restrict__ rstd,          // [M_TOTAL]
    const float* __restrict__ ln_g,          // [HID]
    const float* __restrict__ ln_b,          // [HID]
    float* __restrict__ out)                 // [R][4][OUTD] f32
{
    __shared__ __align__(16) unsigned char lds[2 * MT * LDSPITCH];
    unsigned char* ldsA = lds;
    unsigned char* ldsB = lds + MT * LDSPITCH;

    const int tid  = threadIdx.x;
    const int lane = tid & 31;
    const int wv   = tid >> 5;
    const int wm   = wv & 3;
    const int wn   = wv >> 2;
    const int half = lane >> 4;
    const int l16  = lane & 15;

    const int bm = blockIdx.x;
    const int g  = bm / (R_ROWS / MT);
    const int r0 = (bm % (R_ROWS / MT)) * MT;
    const int n0 = blockIdx.y * NT;              // 0..3 tiles of OUTD=512

    v8f acc[2][4];
    #pragma unroll
    for (int a = 0; a < 2; a++)
        #pragma unroll
        for (int b = 0; b < 4; b++)
            #pragma unroll
            for (int e = 0; e < 8; e++) acc[a][b][e] = 0.0f;

    for (int kt = 0; kt < KDIM / KT; ++kt) {
        const int k0 = kt * KT;
        const int d  = k0 >> 11;
        const int j0 = k0 & (HID - 1);
        const int hs = (g + d) & 3;

        // ---- stage B (async to LDS when available) ----
        #pragma unroll
        for (int c = 0; c < 2; c++) {
            const int chunk = tid * 2 + c;
            const int col   = chunk >> 2;
            const int cp    = chunk & 3;
            const unsigned short* src =
                w3t + (size_t)(n0 + col) * KDIM + k0 + cp * 8;
#if HAVE_ASYNC_LDS
            async_b128(src, ldsB + col * LDSPITCH + cp * 16);
#else
            *(uint4*)(ldsB + col * LDSPITCH + cp * 16) = *(const uint4*)src;
#endif
            if (kt + 1 < KDIM / KT)
                __builtin_prefetch(src + KT, 0, 1);
        }

        // ---- stage A with fused LayerNorm + ReLU (VALU transform) ----
        #pragma unroll
        for (int c = 0; c < 2; c++) {
            const int chunk = tid * 2 + c;
            const int row   = chunk >> 2;
            const int cp    = chunk & 3;
            const int r     = r0 + row;
            const int srow  = r * 4 + hs;
            const float m   = mean[srow];
            const float s   = rstd[srow];
            const unsigned short* src =
                h2 + (size_t)r * KDIM + hs * HID + j0 + cp * 8;
            uint4 raw = *(const uint4*)src;
            const unsigned short* rp = (const unsigned short*)&raw;
            const float4 g0 = *(const float4*)(ln_g + j0 + cp * 8);
            const float4 g1 = *(const float4*)(ln_g + j0 + cp * 8 + 4);
            const float4 bb0 = *(const float4*)(ln_b + j0 + cp * 8);
            const float4 bb1 = *(const float4*)(ln_b + j0 + cp * 8 + 4);
            const float gg[8] = {g0.x, g0.y, g0.z, g0.w, g1.x, g1.y, g1.z, g1.w};
            const float bg[8] = {bb0.x, bb0.y, bb0.z, bb0.w, bb1.x, bb1.y, bb1.z, bb1.w};
            unsigned short op[8];
            #pragma unroll
            for (int e = 0; e < 8; e++) {
                float y = (bf2f(rp[e]) - m) * s * gg[e] + bg[e];
                op[e] = f2bf(fmaxf(y, 0.0f));
            }
            *(uint4*)(ldsA + row * LDSPITCH + cp * 16) = *(const uint4*)op;
        }
#if HAVE_ASYNC_LDS
        wait_asynccnt0();
#endif
        __syncthreads();

        Frag af[2], bfr[4];
        #pragma unroll
        for (int mi = 0; mi < 2; mi++) {
            const int row = wm * 32 + mi * 16 + l16;
            const unsigned char* p = ldsA + row * LDSPITCH + half * 16;
            af[mi].u[0] = *(const uint4*)(p);
            af[mi].u[1] = *(const uint4*)(p + 32);
        }
        #pragma unroll
        for (int ni = 0; ni < 4; ni++) {
            const int col = wn * 64 + ni * 16 + l16;
            const unsigned char* p = ldsB + col * LDSPITCH + half * 16;
            bfr[ni].u[0] = *(const uint4*)(p);
            bfr[ni].u[1] = *(const uint4*)(p + 32);
        }
        #pragma unroll
        for (int mi = 0; mi < 2; mi++)
            #pragma unroll
            for (int ni = 0; ni < 4; ni++)
                acc[mi][ni] = __builtin_amdgcn_wmma_f32_16x16x32_bf16(
                    false, af[mi].v, false, bfr[ni].v,
                    (short)0, acc[mi][ni], false, false);
        __syncthreads();
    }

    // ---- epilogue: + bias, f32 out[r*2048 + g*512 + col] ----
    #pragma unroll
    for (int ni = 0; ni < 4; ni++) {
        const int col  = n0 + wn * 64 + ni * 16 + l16;
        const float bias = b3[col];
        #pragma unroll
        for (int mi = 0; mi < 2; mi++) {
            const int mbase = wm * 32 + mi * 16 + half * 8;
            #pragma unroll
            for (int v = 0; v < 8; v++) {
                const int r = r0 + mbase + v;
                out[(size_t)r * (4 * OUTD) + g * OUTD + col] =
                    acc[mi][ni][v] + bias;
            }
        }
    }
}

// ---------------------------------------------------------------------------
// Launch
// ---------------------------------------------------------------------------
extern "C" void kernel_launch(void* const* d_in, const int* in_sizes, int n_in,
                              void* d_out, int out_size, void* d_ws, size_t ws_size,
                              hipStream_t stream) {
    (void)in_sizes; (void)n_in; (void)out_size; (void)ws_size;
    const float* ins  = (const float*)d_in[0];
    const float* w1   = (const float*)d_in[1];
    const float* b1   = (const float*)d_in[2];
    const float* w2   = (const float*)d_in[3];
    const float* b2   = (const float*)d_in[4];
    const float* w3   = (const float*)d_in[5];
    const float* b3   = (const float*)d_in[6];
    const float* ln_g = (const float*)d_in[7];
    const float* ln_b = (const float*)d_in[8];
    float* out = (float*)d_out;

    uint8_t* ws = (uint8_t*)d_ws;
    unsigned short* w2t = (unsigned short*)(ws);                    //  33,554,432 B
    unsigned short* w3t = (unsigned short*)(ws + 33554432u);        //   8,388,608 B
    unsigned short* h1  = (unsigned short*)(ws + 41943040u);        // 167,772,160 B
    unsigned short* h2  = (unsigned short*)(ws + 209715200u);       // 167,772,160 B
    float* meanp        = (float*)(ws + 377487360u);                //     163,840 B
    float* rstdp        = (float*)(ws + 377651200u);                //     163,840 B

    prep_w2_kernel<<<65536, 256, 0, stream>>>(w2, w2t);
    prep_w3_kernel<<<16384, 256, 0, stream>>>(w3, w3t);
    layer1_kernel<<<R_ROWS / 16, 256, 0, stream>>>(ins, w1, b1, h1);
    gemm2_kernel<<<dim3(M_TOTAL / MT, HID / NT), 256, 0, stream>>>(h1, w2t, b2, h2);
    ln_stats_kernel<<<M_TOTAL, 256, 0, stream>>>(h2, meanp, rstdp);
    gemm3_kernel<<<dim3(M_TOTAL / MT, OUTD / NT), 256, 0, stream>>>(
        h2, w3t, b3, meanp, rstdp, ln_g, ln_b, out);
}
